// DecoderRNN_1632087572899
// MI455X (gfx1250) — matrix operand order repaired and checked by reading, compiled, not run on hardware
//
#include <hip/hip_runtime.h>
#include <math.h>

typedef __attribute__((ext_vector_type(8)))  __bf16 v8bf;
typedef __attribute__((ext_vector_type(16))) __bf16 v16bf;
typedef __attribute__((ext_vector_type(8)))  float  v8f;
typedef __attribute__((ext_vector_type(4)))  unsigned int u32x4;
typedef __attribute__((ext_vector_type(8)))  int i32x8;
typedef __attribute__((ext_vector_type(4)))  int i32x4;

// ---------------------------------------------------------------- constants
#define BB   1024          // batch
#define SS   40            // encoder seq len
#define HH   1024          // hidden
#define DW   512           // embedding width
#define VV   4000          // vocab
#define KCAT (HH + DW)     // 1536
#define H3   (3 * HH)      // 3072

// TDM path: gfx1250 device pass with the clang-23 6-arg tensor builtin.
#if defined(__gfx1250__) && __has_builtin(__builtin_amdgcn_tensor_load_to_lds) && \
    __has_builtin(__builtin_amdgcn_s_wait_tensorcnt)
#define GEMM_USE_TDM 1
#else
#define GEMM_USE_TDM 0
#endif

// ---------------------------------------------------------------- helpers
__global__ void cast_f32_to_bf16(__bf16* __restrict__ dst,
                                 const float* __restrict__ src, size_t n) {
  size_t i = (size_t)blockIdx.x * blockDim.x + threadIdx.x;
  size_t st = (size_t)gridDim.x * blockDim.x;
  for (; i < n; i += st) dst[i] = (__bf16)src[i];
}

__global__ void copy_f32(float* __restrict__ dst,
                         const float* __restrict__ src, size_t n) {
  size_t i = (size_t)blockIdx.x * blockDim.x + threadIdx.x;
  size_t st = (size_t)gridDim.x * blockDim.x;
  for (; i < n; i += st) dst[i] = src[i];
}

// load 16 bf16 as two contiguous 8-element runs separated by `gap` elements
static __device__ __forceinline__ v16bf ld16(const __bf16* p, int gap) {
  v8bf lo = *(const v8bf*)p;
  v8bf hi = *(const v8bf*)(p + gap);
  v16bf r;
#pragma unroll
  for (int i = 0; i < 8; ++i) { r[i] = lo[i]; r[8 + i] = hi[i]; }
  return r;
}

#if GEMM_USE_TDM
// Issue one TDM 2D tile load: 32(K) x 128(rows) bf16 tile of W into LDS.
// D# built per CDNA5 ISA ch.8: group0 = {count, lds_addr, global_addr, type=2},
// group1 = {data_size=2B, tensor_dim0/1 (remaining K / remaining rows, so OOB
// rows are zero-filled), tile_dim0=32, tile_dim1=128, dim0 stride = ldw}.
static __device__ __forceinline__ void tdm_load_w_tile(
    const __bf16* gptr, unsigned lds_byte_addr, int k_rem, int n_rem, int ldw) {
  unsigned long long ga = (unsigned long long)(uintptr_t)gptr;
  unsigned td0 = (unsigned)k_rem;
  unsigned td1 = (unsigned)(n_rem > 0 ? n_rem : 1);
  u32x4 g0; i32x8 g1; i32x4 g2; i32x4 g3; i32x8 g4;
  g0[0] = 1u;                                   // count=1, no gather
  g0[1] = lds_byte_addr;                        // LDS byte address
  g0[2] = (unsigned)(ga & 0xffffffffu);         // global_addr[31:0]
  g0[3] = (unsigned)((ga >> 32) & 0x01ffffffu)  // global_addr[56:32]
        | (2u << 30);                           // type=2 ("image")
  g1[0] = (int)(1u << 16);                      // data_size=1 (2 bytes)
  g1[1] = (int)((td0 & 0xffffu) << 16);         // tensor_dim0[15:0]
  g1[2] = (int)((td0 >> 16) | ((td1 & 0xffffu) << 16)); // td0 hi | td1 lo
  g1[3] = (int)((td1 >> 16) | (32u << 16));     // td1 hi | tile_dim0=32
  g1[4] = (int)128u;                            // tile_dim1=128, tile_dim2=0
  g1[5] = (int)(unsigned)ldw;                   // dim0 stride [31:0]
  g1[6] = 0;                                    // dim0 stride hi, dim1 stride lo
  g1[7] = 0;
  g2[0] = g2[1] = g2[2] = g2[3] = 0;
  g3[0] = g3[1] = g3[2] = g3[3] = 0;
#pragma unroll
  for (int i = 0; i < 8; ++i) g4[i] = 0;
  __builtin_amdgcn_tensor_load_to_lds(g0, g1, g2, g3, g4, 0);
}
#endif

// ---------------------------------------------------------------- WMMA GEMM
// C[M x N] = A[M x K](bf16, row-major, lda) * W[N x K]^T (bf16, row-major, ldw)
// M multiple of 64; grid = (ceil(N/128), M/64); block = 256 (8 waves, 2x4 of
// 32x32 wave tiles). W tiles are staged through LDS by the Tensor Data Mover
// (double-buffered, TENSORcnt-gated); A streams directly from global.
__global__ __launch_bounds__(256) void gemm_bf16_nt(
    const __bf16* __restrict__ A, int lda,
    const __bf16* __restrict__ W, int ldw,
    float* __restrict__ Cf, __bf16* __restrict__ Cb, int ldc,
    int Nvalid, int K) {
  const int lane = threadIdx.x & 31;
  const int wave = threadIdx.x >> 5;
  const int m0 = blockIdx.y * 64 + (wave >> 2) * 32;
  const int n0 = blockIdx.x * 128 + (wave & 3) * 32;
  const int r  = lane & 15;   // row (A) / col (B,C) within 16-tile
  const int kg = lane >> 4;   // K-half select

  v8f acc[2][2];
#pragma unroll
  for (int i = 0; i < 2; ++i)
#pragma unroll
    for (int j = 0; j < 2; ++j)
#pragma unroll
      for (int v = 0; v < 8; ++v) acc[i][j][v] = 0.0f;

  // A: lane holds row m, K = kg*8..+8 and 16+kg*8..+8 of each 32-slice
  const __bf16* a0 = A + (size_t)(m0 + r) * lda + kg * 8;
  const __bf16* a1 = a0 + (size_t)16 * lda;

#if GEMM_USE_TDM
  __shared__ __bf16 lds_w[2][128 * 32];   // two 8 KB W tiles (rows x K-slice)
  const int nsteps = K >> 5;
  const int nb = blockIdx.x * 128;                  // block's W row base
  const int n_rem = Nvalid - nb;                    // rows available (>0)
  const __bf16* wt = W + (size_t)nb * ldw;          // tile origin (k=0)
  const unsigned lb0 = (unsigned)(uintptr_t)&lds_w[0][0];
  const unsigned lb1 = (unsigned)(uintptr_t)&lds_w[1][0];
  // lane's W rows inside the staged tile
  const int nl0 = (wave & 3) * 32 + r;
  const int nl1 = nl0 + 16;

  if (wave == 0) tdm_load_w_tile(wt, lb0, K, n_rem, ldw);

  for (int i = 0; i < nsteps; ++i) {
    if (wave == 0) __builtin_amdgcn_s_wait_tensorcnt(0);
    __syncthreads();                      // current tile visible to all waves
    if (wave == 0 && i + 1 < nsteps)      // overlap next DMA with compute
      tdm_load_w_tile(wt + (size_t)(i + 1) * 32, (i & 1) ? lb0 : lb1,
                      K - (i + 1) * 32, n_rem, ldw);
    const int k0 = i * 32;
    __builtin_prefetch(a0 + k0 + 128, 0, 1);   // global_prefetch_b8 (A stream)
    const __bf16* lw = &lds_w[i & 1][0];
    v16bf af0 = ld16(a0 + k0, 16);
    v16bf af1 = ld16(a1 + k0, 16);
    v16bf bf0 = ld16(lw + nl0 * 32 + kg * 8, 16);   // ds_load_b128 x2
    v16bf bf1 = ld16(lw + nl1 * 32 + kg * 8, 16);
    acc[0][0] = __builtin_amdgcn_wmma_f32_16x16x32_bf16(false, af0, false, bf0,
                                                        (short)0, acc[0][0], false, false);
    acc[0][1] = __builtin_amdgcn_wmma_f32_16x16x32_bf16(false, af0, false, bf1,
                                                        (short)0, acc[0][1], false, false);
    acc[1][0] = __builtin_amdgcn_wmma_f32_16x16x32_bf16(false, af1, false, bf0,
                                                        (short)0, acc[1][0], false, false);
    acc[1][1] = __builtin_amdgcn_wmma_f32_16x16x32_bf16(false, af1, false, bf1,
                                                        (short)0, acc[1][1], false, false);
    __syncthreads();                      // all reads done before tile reuse
  }
#else
  // Fallback: direct-global W fragments
  int wr0 = n0 + r;      if (wr0 >= Nvalid) wr0 = Nvalid - 1;
  int wr1 = n0 + 16 + r; if (wr1 >= Nvalid) wr1 = Nvalid - 1;
  const __bf16* w0 = W + (size_t)wr0 * ldw + kg * 8;
  const __bf16* w1 = W + (size_t)wr1 * ldw + kg * 8;
  for (int k0 = 0; k0 < K; k0 += 32) {
    __builtin_prefetch(a0 + k0 + 128, 0, 1);
    __builtin_prefetch(w0 + k0 + 128, 0, 1);
    v16bf af0 = ld16(a0 + k0, 16);
    v16bf af1 = ld16(a1 + k0, 16);
    v16bf bf0 = ld16(w0 + k0, 16);
    v16bf bf1 = ld16(w1 + k0, 16);
    acc[0][0] = __builtin_amdgcn_wmma_f32_16x16x32_bf16(false, af0, false, bf0,
                                                        (short)0, acc[0][0], false, false);
    acc[0][1] = __builtin_amdgcn_wmma_f32_16x16x32_bf16(false, af0, false, bf1,
                                                        (short)0, acc[0][1], false, false);
    acc[1][0] = __builtin_amdgcn_wmma_f32_16x16x32_bf16(false, af1, false, bf0,
                                                        (short)0, acc[1][0], false, false);
    acc[1][1] = __builtin_amdgcn_wmma_f32_16x16x32_bf16(false, af1, false, bf1,
                                                        (short)0, acc[1][1], false, false);
  }
#endif

  const int cr = 8 * (lane >> 4);   // C row offset from lane-half
#pragma unroll
  for (int mi = 0; mi < 2; ++mi)
#pragma unroll
    for (int ni = 0; ni < 2; ++ni) {
      int col = n0 + ni * 16 + r;
      if (col < Nvalid) {
        size_t base = (size_t)(m0 + mi * 16 + cr) * ldc + col;
#pragma unroll
        for (int v = 0; v < 8; ++v) {
          size_t idx = base + (size_t)v * ldc;
          if (Cb) Cb[idx] = (__bf16)acc[mi][ni][v];
          else    Cf[idx] = acc[mi][ni][v];
        }
      }
    }
}

// ------------------------------------------------- fused tanh + w2 dot score
// e[b,s] = sum_h tanh(encpart[b,s,h] + hpart[b,h] + b1[h]) * w2[h]
__global__ __launch_bounds__(256) void attn_score(
    const __bf16* __restrict__ encpart, const float* __restrict__ hpart,
    const float* __restrict__ b1, const float* __restrict__ w2,
    float* __restrict__ e) {
  const int bs = blockIdx.x;            // 0 .. B*S-1
  const int b  = bs / SS;
  __shared__ float red[256];
  float sum = 0.0f;
  for (int h = threadIdx.x; h < HH; h += 256) {
    float v = (float)encpart[(size_t)bs * HH + h] + hpart[(size_t)b * HH + h] + b1[h];
    sum += tanhf(v) * w2[h];
  }
  red[threadIdx.x] = sum;
  __syncthreads();
  for (int off = 128; off > 0; off >>= 1) {
    if (threadIdx.x < off) red[threadIdx.x] += red[threadIdx.x + off];
    __syncthreads();
  }
  if (threadIdx.x == 0) e[bs] = red[0];
}

// softmax over S=40, one thread per batch row
__global__ void softmax_rows(const float* __restrict__ e,
                             float* __restrict__ alpha) {
  int b = blockIdx.x * blockDim.x + threadIdx.x;
  if (b >= BB) return;
  const float* p = e + (size_t)b * SS;
  float mx = -3.4e38f;
  for (int s = 0; s < SS; ++s) mx = fmaxf(mx, p[s]);
  float sum = 0.0f;
  for (int s = 0; s < SS; ++s) sum += expf(p[s] - mx);
  float inv = 1.0f / sum;
  for (int s = 0; s < SS; ++s) alpha[(size_t)b * SS + s] = expf(p[s] - mx) * inv;
}

// context[b,h] = sum_s alpha[b,s] * enc[b,s,h]
__global__ __launch_bounds__(256) void context_kernel(
    const float* __restrict__ alpha, const float* __restrict__ enc,
    float* __restrict__ ctx) {
  size_t idx = (size_t)blockIdx.x * blockDim.x + threadIdx.x;
  if (idx >= (size_t)BB * HH) return;
  int b = (int)(idx / HH), h = (int)(idx % HH);
  float s = 0.0f;
  for (int si = 0; si < SS; ++si)
    s += alpha[(size_t)b * SS + si] * enc[((size_t)b * SS + si) * HH + h];
  ctx[idx] = s;
}

// xcat[b, 0..DW) = bf16(emb[it[b]]), xcat[b, DW..) = bf16(context[b])
__global__ __launch_bounds__(256) void build_x(
    const float* __restrict__ emb, const int* __restrict__ targets,
    const float* __restrict__ ctx, __bf16* __restrict__ xcat, int t) {
  size_t idx = (size_t)blockIdx.x * blockDim.x + threadIdx.x;
  if (idx >= (size_t)BB * KCAT) return;
  int b = (int)(idx / KCAT), c = (int)(idx % KCAT);
  if (c < DW) {
    int it = 0;
    if (t == 1)      it = targets[b] + 1;           // targets[0, b] + 1
    else if (t == 2) it = targets[BB + b] + 36;     // targets[1, b] + 36
    xcat[idx] = (__bf16)emb[(size_t)it * DW + c];
  } else {
    xcat[idx] = (__bf16)ctx[(size_t)b * HH + (c - DW)];
  }
}

// GRU gate fusion; h updated in place (pure elementwise in j)
__global__ __launch_bounds__(256) void gru_gates(
    const float* __restrict__ gi, const float* __restrict__ gh,
    const float* __restrict__ bih, const float* __restrict__ bhh,
    float* __restrict__ h) {
  size_t idx = (size_t)blockIdx.x * blockDim.x + threadIdx.x;
  if (idx >= (size_t)BB * HH) return;
  int b = (int)(idx / HH), j = (int)(idx % HH);
  size_t base = (size_t)b * H3;
  float ir = gi[base + j]          + bih[j];
  float iz = gi[base + HH + j]     + bih[HH + j];
  float in = gi[base + 2 * HH + j] + bih[2 * HH + j];
  float hr = gh[base + j]          + bhh[j];
  float hz = gh[base + HH + j]     + bhh[HH + j];
  float hn = gh[base + 2 * HH + j] + bhh[2 * HH + j];
  float rg = 1.0f / (1.0f + expf(-(ir + hr)));
  float zg = 1.0f / (1.0f + expf(-(iz + hz)));
  float ng = tanhf(in + rg * hn);
  h[idx] = (1.0f - zg) * ng + zg * h[idx];
}

// in-place log_softmax over V=4000 with bias add; one block per row
__global__ __launch_bounds__(256) void log_softmax_rows(
    float* __restrict__ logits, const float* __restrict__ bias) {
  float* p = logits + (size_t)blockIdx.x * VV;
  __shared__ float red[256];
  float mx = -3.4e38f;
  for (int j = threadIdx.x; j < VV; j += 256) mx = fmaxf(mx, p[j] + bias[j]);
  red[threadIdx.x] = mx;
  __syncthreads();
  for (int off = 128; off > 0; off >>= 1) {
    if (threadIdx.x < off) red[threadIdx.x] = fmaxf(red[threadIdx.x], red[threadIdx.x + off]);
    __syncthreads();
  }
  mx = red[0];
  __syncthreads();
  float sum = 0.0f;
  for (int j = threadIdx.x; j < VV; j += 256) sum += expf(p[j] + bias[j] - mx);
  red[threadIdx.x] = sum;
  __syncthreads();
  for (int off = 128; off > 0; off >>= 1) {
    if (threadIdx.x < off) red[threadIdx.x] += red[threadIdx.x + off];
    __syncthreads();
  }
  float lse = mx + logf(red[0]);
  for (int j = threadIdx.x; j < VV; j += 256) p[j] = p[j] + bias[j] - lse;
}

// ---------------------------------------------------------------- launcher
extern "C" void kernel_launch(void* const* d_in, const int* in_sizes, int n_in,
                              void* d_out, int out_size, void* d_ws, size_t ws_size,
                              hipStream_t stream) {
  (void)in_sizes; (void)n_in; (void)out_size; (void)ws_size;
  const float* enc   = (const float*)d_in[0];   // (B,S,H)
  const float* h0    = (const float*)d_in[1];   // (1,B,H)
  const int*   tgt   = (const int*)  d_in[2];   // (3,B)
  const float* emb   = (const float*)d_in[3];   // (8001,DW)
  const float* w1    = (const float*)d_in[4];   // (H,2H)
  const float* b1    = (const float*)d_in[5];   // (H)
  const float* w2    = (const float*)d_in[6];   // (1,H)
  const float* wih   = (const float*)d_in[7];   // (3H, H+DW)
  const float* whh   = (const float*)d_in[8];   // (3H, H)
  const float* bih   = (const float*)d_in[9];
  const float* bhh   = (const float*)d_in[10];
  const float* o0w   = (const float*)d_in[11];  // (V,H)
  const float* o0b   = (const float*)d_in[12];
  const float* o1w   = (const float*)d_in[13];
  const float* o1b   = (const float*)d_in[14];
  float* out = (float*)d_out;                   // (3,B,V)

  // ---- carve workspace
  size_t off = 0;
  auto carve = [&](size_t bytes) -> void* {
    void* p = (char*)d_ws + off;
    off += (bytes + 255) & ~(size_t)255;
    return p;
  };
  const size_t BS = (size_t)BB * SS;
  __bf16* enc_b   = (__bf16*)carve(BS * HH * 2);
  __bf16* encpart = (__bf16*)carve(BS * HH * 2);         // enc @ w1a.T (bf16)
  __bf16* w1_b    = (__bf16*)carve((size_t)HH * 2 * HH * 2);
  __bf16* wih_b   = (__bf16*)carve((size_t)H3 * KCAT * 2);
  __bf16* whh_b   = (__bf16*)carve((size_t)H3 * HH * 2);
  __bf16* o0_b    = (__bf16*)carve((size_t)VV * HH * 2);
  __bf16* o1_b    = (__bf16*)carve((size_t)VV * HH * 2);
  float*  h_f     = (float*) carve((size_t)BB * HH * 4);
  __bf16* h_b     = (__bf16*)carve((size_t)BB * HH * 2);
  float*  hpart   = (float*) carve((size_t)BB * HH * 4);
  float*  evec    = (float*) carve(BS * 4);
  float*  alpha   = (float*) carve(BS * 4);
  float*  ctx     = (float*) carve((size_t)BB * HH * 4);
  __bf16* xcat    = (__bf16*)carve((size_t)BB * KCAT * 2);
  float*  gi      = (float*) carve((size_t)BB * H3 * 4);
  float*  gh      = (float*) carve((size_t)BB * H3 * 4);

  auto cast = [&](__bf16* dst, const float* src, size_t n) {
    int blocks = (int)((n + 255) / 256); if (blocks > 8192) blocks = 8192;
    cast_f32_to_bf16<<<blocks, 256, 0, stream>>>(dst, src, n);
  };

  // ---- one-time precompute
  cast(enc_b, enc, BS * HH);
  cast(w1_b,  w1,  (size_t)HH * 2 * HH);
  cast(wih_b, wih, (size_t)H3 * KCAT);
  cast(whh_b, whh, (size_t)H3 * HH);
  cast(o0_b,  o0w, (size_t)VV * HH);
  cast(o1_b,  o1w, (size_t)VV * HH);
  copy_f32<<<4096, 256, 0, stream>>>(h_f, h0, (size_t)BB * HH);

  // encpart = enc @ w1[:, :H].T   (M=40960, N=1024, K=1024) -> bf16
  gemm_bf16_nt<<<dim3(HH / 128, (unsigned)(BS / 64)), 256, 0, stream>>>(
      enc_b, HH, w1_b, 2 * HH, nullptr, encpart, HH, HH, HH);

  // ---- decode loop
  for (int t = 0; t < 3; ++t) {
    cast(h_b, h_f, (size_t)BB * HH);

    // hpart = h @ w1[:, H:].T   (1024 x 1024 x 1024)
    gemm_bf16_nt<<<dim3(HH / 128, BB / 64), 256, 0, stream>>>(
        h_b, HH, w1_b + HH, 2 * HH, hpart, nullptr, HH, HH, HH);

    attn_score<<<(unsigned)BS, 256, 0, stream>>>(encpart, hpart, b1, w2, evec);
    softmax_rows<<<(BB + 255) / 256, 256, 0, stream>>>(evec, alpha);
    context_kernel<<<(unsigned)((size_t)BB * HH / 256), 256, 0, stream>>>(alpha, enc, ctx);
    build_x<<<(unsigned)(((size_t)BB * KCAT + 255) / 256), 256, 0, stream>>>(
        emb, tgt, ctx, xcat, t);

    // gi = xcat @ w_ih.T (1024 x 3072 x 1536); gh = h @ w_hh.T (1024 x 3072 x 1024)
    gemm_bf16_nt<<<dim3(H3 / 128, BB / 64), 256, 0, stream>>>(
        xcat, KCAT, wih_b, KCAT, gi, nullptr, H3, H3, KCAT);
    gemm_bf16_nt<<<dim3(H3 / 128, BB / 64), 256, 0, stream>>>(
        h_b, HH, whh_b, HH, gh, nullptr, H3, H3, HH);

    gru_gates<<<(unsigned)((size_t)BB * HH / 256), 256, 0, stream>>>(gi, gh, bih, bhh, h_f);

    // logits = h @ out_w.T (1024 x 4000 x 1024), written straight into d_out
    cast(h_b, h_f, (size_t)BB * HH);
    const __bf16* ow = (t == 1) ? o1_b : o0_b;
    const float*  ob = (t == 1) ? o1b  : o0b;
    float* lt = out + (size_t)t * BB * VV;
    gemm_bf16_nt<<<dim3((VV + 127) / 128, BB / 64), 256, 0, stream>>>(
        h_b, HH, ow, HH, lt, nullptr, VV, VV, HH);
    log_softmax_rows<<<BB, 256, 0, stream>>>(lt, ob);
  }
}
